// Attention_8349416424073
// MI455X (gfx1250) — matrix operand order repaired and checked by reading, not compiled, not run on hardware
//
#include <hip/hip_runtime.h>
#include <hip/hip_bf16.h>
#include <math.h>

typedef __bf16 bf16;
typedef __attribute__((ext_vector_type(8)))  bf16  v8bf;
typedef __attribute__((ext_vector_type(16))) bf16  v16bf;
typedef __attribute__((ext_vector_type(8)))  float v8f;

#define WMMA_BF16(a, b, c) \
  __builtin_amdgcn_wmma_f32_16x16x32_bf16(false, (a), false, (b), (short)0, (c), false, false)

// Load a 16-element bf16 A/B fragment from two (possibly disjoint) 16-byte chunks.
__device__ __forceinline__ v16bf ld_pair(const bf16* p0, const bf16* p1) {
  v8bf lo = *reinterpret_cast<const v8bf*>(p0);
  v8bf hi = *reinterpret_cast<const v8bf*>(p1);
  return __builtin_shufflevector(lo, hi, 0,1,2,3,4,5,6,7,8,9,10,11,12,13,14,15);
}

// Async copy of 16 bytes/lane from global to LDS (CDNA5 ASYNCcnt path).
__device__ __forceinline__ void async_b128_to_lds(const bf16* src, bf16* lds_dst) {
  unsigned ldsoff = (unsigned)(uintptr_t)lds_dst;   // flat LDS addr: low 32 bits = LDS offset
  asm volatile("global_load_async_to_lds_b128 %0, %1, off"
               :: "v"(ldsoff), "v"(src)
               : "memory");
}

// ---------------------------------------------------------------- elementwise
__global__ void cvt_f32_to_bf16(const float* __restrict__ in, bf16* __restrict__ out, int n) {
  int i = blockIdx.x * 256 + threadIdx.x;
  if (i < n) out[i] = (bf16)in[i];
}

// Wt[n][k] = (bf16) W[k][n], 512x512
__global__ void cvt_transpose_w(const float* __restrict__ W, bf16* __restrict__ Wt) {
  int i = blockIdx.x * 256 + threadIdx.x;          // over 512*512
  int n = i >> 9, k = i & 511;
  Wt[(size_t)n * 512 + k] = (bf16)W[(size_t)k * 512 + n];
}

__global__ void gate_mul(const float* __restrict__ O, const float* __restrict__ G,
                         bf16* __restrict__ out, int n) {
  int i = blockIdx.x * 256 + threadIdx.x;
  if (i < n) out[i] = (bf16)(O[i] * G[i]);
}

// ------------------------------------------------- inception depthwise conv
// X: fp32 (64, 1024, 64) split-head layout. sh<32 passthrough, sh>=32 conv along L.
// scale applied to everything (q: D^-0.5). transpose_out: emit (64, 64, 1024) d-major.
__global__ void dwconv_inception(const float* __restrict__ X, const float* __restrict__ w,
                                 const float* __restrict__ b, bf16* __restrict__ out,
                                 float scale, int transpose_out) {
  int i  = blockIdx.x * 256 + threadIdx.x;         // 64*1024*64 total
  int d  = i & 63;
  int l  = (i >> 6) & 1023;
  int sh = i >> 16;
  float y;
  if (sh < 32) {
    y = X[i];
  } else {
    float xm = (l > 0)    ? X[i - 64] : 0.f;
    float x0 = X[i];
    float xp = (l < 1023) ? X[i + 64] : 0.f;
    y = w[d * 3 + 0] * xm + w[d * 3 + 1] * x0 + w[d * 3 + 2] * xp + b[d];
  }
  y *= scale;
  if (transpose_out) out[((size_t)sh * 64 + d) * 1024 + l] = (bf16)y;
  else               out[i] = (bf16)y;
}

// ------------------------------------------------------------ WMMA GEMM
// C(M,N) f32 = A(M,K) bf16 row-major  x  Bt(N,K) bf16 row-major (pre-transposed B).
// B tile (64 cols x 32 k) is staged per-block into LDS via async global->LDS copies
// (double buffered, ASYNCcnt-synchronized) and shared by all 8 waves.
// mode 0: split-head store  (m=s*1024+l, n=h*64+d) -> [(s*8+h)*1024+l]*64+d
// mode 1: sigmoid(x + bias[n]) plain store
// mode 2: x + bias[n] plain store
__global__ __launch_bounds__(256) void gemm_bf16_wmma(
    const bf16* __restrict__ A, const bf16* __restrict__ Bt,
    float* __restrict__ C, const float* __restrict__ bias,
    int M, int N, int K, int mode) {
  __shared__ alignas(16) bf16 Bs[2][64][32];       // 8 KB double buffer
  const int lane = threadIdx.x & 31;
  const int wave = threadIdx.x >> 5;
  const int half = lane >> 4;
  const int l15  = lane & 15;
  const int row0 = blockIdx.y * 128 + wave * 16;
  const int col0 = blockIdx.x * 64;
  const int srow = threadIdx.x >> 2;               // staging: B row 0..63
  const int schk = (threadIdx.x & 3) * 8;          // staging: 8-elem chunk in k

  const v8f zero = {0.f, 0.f, 0.f, 0.f, 0.f, 0.f, 0.f, 0.f};
  v8f acc[4] = {zero, zero, zero, zero};

  const bf16* aptr  = A + (size_t)(row0 + l15) * K;
  const bf16* bstag = Bt + (size_t)(col0 + srow) * K + schk;
  const int nsteps = K / 32;

  // prologue: stage k-step 0 into buffer 0
  async_b128_to_lds(bstag, &Bs[0][srow][schk]);

  int buf = 0;
  for (int i = 0; i < nsteps; ++i) {
    const int kb = i * 32;
    if (i + 1 < nsteps) {
      async_b128_to_lds(bstag + kb + 32, &Bs[buf ^ 1][srow][schk]);
      asm volatile("s_wait_asynccnt 0x1" ::: "memory");  // my stage i landed
    } else {
      asm volatile("s_wait_asynccnt 0x0" ::: "memory");
    }
    __syncthreads();                               // all waves' stage i visible

    __builtin_prefetch(aptr + kb + 128, 0, 0);
    // A fragment: k(e) = 8*half + (e&7) + 16*(e>>3)
    v16bf af = ld_pair(aptr + kb + 8 * half, aptr + kb + 16 + 8 * half);
    #pragma unroll
    for (int t = 0; t < 4; ++t) {
      // B fragment from LDS: k(e) = 8*half + (e&7) + 16*(e>>3)
      const bf16* bptr = &Bs[buf][t * 16 + l15][8 * half];
      v16bf bfrag = ld_pair(bptr, bptr + 16);
      acc[t] = WMMA_BF16(af, bfrag, acc[t]);
    }
    __syncthreads();                               // reads done before buf is overwritten
    buf ^= 1;
  }

  #pragma unroll
  for (int t = 0; t < 4; ++t) {
    #pragma unroll
    for (int r = 0; r < 8; ++r) {
      int rowo = row0 + r + 8 * half;             // C layout: M = r + 8*(lane>>4)
      int colo = col0 + t * 16 + l15;             // N = lane&15
      float v = acc[t][r];
      if (mode == 0) {
        int s_ = rowo >> 10, l_ = rowo & 1023;
        int h_ = colo >> 6,  d_ = colo & 63;
        C[(((size_t)s_ * 8 + h_) * 1024 + l_) * 64 + d_] = v;
      } else if (mode == 1) {
        v += bias[colo];
        C[(size_t)rowo * N + colo] = 1.0f / (1.0f + __expf(-v));
      } else {
        C[(size_t)rowo * N + colo] = v + bias[colo];
      }
    }
  }
}

// ------------------------------------------------------------ fused attention
// Qb,Kb: bf16 (64, 1024, 64); Vbt: bf16 (64, 64, 1024) d-major.
// bias: f32 (8, 1024, 1024) per-head; kmask: int (8, 1024) per-s.
// Of: f32 (8, 1024, 512) = (s, l, h*64+d). One block = one (s,h) pair x 128 queries.
__global__ __launch_bounds__(256) void attn_fused(
    const bf16* __restrict__ Qb, const bf16* __restrict__ Kb, const bf16* __restrict__ Vbt,
    const float* __restrict__ bias, const int* __restrict__ kmask,
    float* __restrict__ Of) {
  __shared__ alignas(16) bf16 pl[8][16][32];       // per-wave P tile (16 q x 32 keys)
  const int lane = threadIdx.x & 31;
  const int wave = threadIdx.x >> 5;
  const int half = lane >> 4;
  const int l15  = lane & 15;
  const int sh = blockIdx.x >> 3;                  // 0..63
  const int qb = blockIdx.x & 7;                   // 0..7
  const int s  = sh >> 3;
  const int h  = sh & 7;
  const int qr0 = qb * 128 + wave * 16;

  // Q A-fragments (held for the whole key loop), contraction dim d = 0..63
  const bf16* qrow = Qb + ((size_t)sh * 1024 + qr0 + l15) * 64;
  v16bf qa0 = ld_pair(qrow + 16 * half,      qrow + 8 + 16 * half);
  v16bf qa1 = ld_pair(qrow + 32 + 16 * half, qrow + 40 + 16 * half);

  const v8f zero = {0.f, 0.f, 0.f, 0.f, 0.f, 0.f, 0.f, 0.f};
  v8f o0 = zero, o1 = zero, o2 = zero, o3 = zero;
  float mrow[8], srow[8];
  #pragma unroll
  for (int r = 0; r < 8; ++r) { mrow[r] = -3.402823466e38f; srow[r] = 0.f; }

  const float* bias_h = bias + (size_t)h * 1024 * 1024;
  const int*   mk_s   = kmask + s * 1024;

  for (int kk = 0; kk < 1024; kk += 32) {
    #pragma unroll
    for (int sub = 0; sub < 2; ++sub) {
      int key = kk + sub * 16 + l15;               // this lane's key column
      const bf16* krow = Kb + ((size_t)sh * 1024 + key) * 64;
      v16bf kf0 = ld_pair(krow + 16 * half,      krow + 8 + 16 * half);   // d 0..31
      v16bf kf1 = ld_pair(krow + 32 + 16 * half, krow + 40 + 16 * half);  // d 32..63
      v8f c = zero;
      c = WMMA_BF16(qa0, kf0, c);
      c = WMMA_BF16(qa1, kf1, c);
      bool msk = (mk_s[key] != 0);
      #pragma unroll
      for (int r = 0; r < 8; ++r) {
        int rq = qr0 + r + 8 * half;
        float lg = msk ? (c[r] + bias_h[(size_t)rq * 1024 + key]) : -3.402823466e38f;
        // row max over the 16-lane group holding this row
        float tm = lg;
        tm = fmaxf(tm, __shfl_xor(tm, 1, 32));
        tm = fmaxf(tm, __shfl_xor(tm, 2, 32));
        tm = fmaxf(tm, __shfl_xor(tm, 4, 32));
        tm = fmaxf(tm, __shfl_xor(tm, 8, 32));
        float mnew  = fmaxf(mrow[r], tm);
        float alpha = __expf(mrow[r] - mnew);
        float pe    = __expf(lg - mnew);
        float ps = pe;
        ps += __shfl_xor(ps, 1, 32);
        ps += __shfl_xor(ps, 2, 32);
        ps += __shfl_xor(ps, 4, 32);
        ps += __shfl_xor(ps, 8, 32);
        srow[r] = srow[r] * alpha + ps;
        mrow[r] = mnew;
        o0[r] *= alpha; o1[r] *= alpha; o2[r] *= alpha; o3[r] *= alpha;
        pl[wave][r + 8 * half][sub * 16 + l15] = (bf16)pe;   // C-layout -> LDS
      }
    }
    asm volatile("s_wait_dscnt 0" ::: "memory");   // P stores visible to whole wave
    // Re-read P in A-fragment layout: row = l15, k(e) = e + 16*half
    const bf16* prow = &pl[wave][l15][0];
    v16bf pa = ld_pair(prow + 16 * half, prow + 16 * half + 8);
    {
      const bf16* vb0 = Vbt + ((size_t)sh * 64 +  0 + l15) * 1024 + kk + 16 * half;
      const bf16* vb1 = Vbt + ((size_t)sh * 64 + 16 + l15) * 1024 + kk + 16 * half;
      const bf16* vb2 = Vbt + ((size_t)sh * 64 + 32 + l15) * 1024 + kk + 16 * half;
      const bf16* vb3 = Vbt + ((size_t)sh * 64 + 48 + l15) * 1024 + kk + 16 * half;
      o0 = WMMA_BF16(pa, ld_pair(vb0, vb0 + 8), o0);
      o1 = WMMA_BF16(pa, ld_pair(vb1, vb1 + 8), o1);
      o2 = WMMA_BF16(pa, ld_pair(vb2, vb2 + 8), o2);
      o3 = WMMA_BF16(pa, ld_pair(vb3, vb3 + 8), o3);
    }
  }

  #pragma unroll
  for (int r = 0; r < 8; ++r) {
    int rq = qr0 + r + 8 * half;
    float inv = 1.0f / srow[r];
    size_t base = ((size_t)s * 1024 + rq) * 512 + h * 64 + l15;
    Of[base +  0] = o0[r] * inv;
    Of[base + 16] = o1[r] * inv;
    Of[base + 32] = o2[r] * inv;
    Of[base + 48] = o3[r] * inv;
  }
}

// ---------------------------------------------------------------- launch
extern "C" void kernel_launch(void* const* d_in, const int* in_sizes, int n_in,
                              void* d_out, int out_size, void* d_ws, size_t ws_size,
                              hipStream_t stream) {
  const float* q_data = (const float*)d_in[0];
  const float* k_data = (const float*)d_in[1];
  const float* bias   = (const float*)d_in[2];
  const int*   k_mask = (const int*)  d_in[3];
  const float* Wq  = (const float*)d_in[4];
  const float* Wk  = (const float*)d_in[5];
  const float* Wv  = (const float*)d_in[6];
  const float* Wg  = (const float*)d_in[7];
  const float* bg  = (const float*)d_in[8];
  const float* Wo  = (const float*)d_in[9];
  const float* bo  = (const float*)d_in[10];
  const float* qcw = (const float*)d_in[11];
  const float* qcb = (const float*)d_in[12];
  const float* kcw = (const float*)d_in[13];
  const float* kcb = (const float*)d_in[14];
  const float* vcw = (const float*)d_in[15];
  const float* vcb = (const float*)d_in[16];
  float* out = (float*)d_out;

  const int M = 8192, N = 512, K = 512;
  const int NELT = M * N;                          // 4,194,304

  char* ws = (char*)d_ws;
  auto alloc = [&](size_t bytes) -> char* {
    char* p = ws; ws += (bytes + 255) & ~(size_t)255; return p;
  };
  bf16* qdb  = (bf16*)alloc((size_t)NELT * 2);
  bf16* kdb  = (bf16*)alloc((size_t)NELT * 2);
  bf16* Wqt  = (bf16*)alloc(512 * 512 * 2);
  bf16* Wkt  = (bf16*)alloc(512 * 512 * 2);
  bf16* Wvt  = (bf16*)alloc(512 * 512 * 2);
  bf16* Wgt  = (bf16*)alloc(512 * 512 * 2);
  bf16* Wot  = (bf16*)alloc(512 * 512 * 2);
  float* Qf  = (float*)alloc((size_t)NELT * 4);
  float* Kf  = (float*)alloc((size_t)NELT * 4);
  float* Vf  = (float*)alloc((size_t)NELT * 4);
  float* G   = (float*)alloc((size_t)NELT * 4);
  bf16* Qb   = (bf16*)alloc((size_t)NELT * 2);
  bf16* Kb   = (bf16*)alloc((size_t)NELT * 2);
  bf16* Vbt  = (bf16*)alloc((size_t)NELT * 2);
  float* Of  = (float*)alloc((size_t)NELT * 4);
  bf16* OGb  = (bf16*)alloc((size_t)NELT * 2);

  const int EB = NELT / 256;                       // elementwise blocks
  dim3 gemmGrid(N / 64, M / 128);

  // 1) convert activations / weights to bf16 (weights transposed)
  cvt_f32_to_bf16<<<EB, 256, 0, stream>>>(q_data, qdb, NELT);
  cvt_f32_to_bf16<<<EB, 256, 0, stream>>>(k_data, kdb, NELT);
  cvt_transpose_w<<<1024, 256, 0, stream>>>(Wq, Wqt);
  cvt_transpose_w<<<1024, 256, 0, stream>>>(Wk, Wkt);
  cvt_transpose_w<<<1024, 256, 0, stream>>>(Wv, Wvt);
  cvt_transpose_w<<<1024, 256, 0, stream>>>(Wg, Wgt);
  cvt_transpose_w<<<1024, 256, 0, stream>>>(Wo, Wot);

  // 2) projections (split-head f32) + gate (sigmoid)
  gemm_bf16_wmma<<<gemmGrid, 256, 0, stream>>>(qdb, Wqt, Qf, nullptr, M, N, K, 0);
  gemm_bf16_wmma<<<gemmGrid, 256, 0, stream>>>(kdb, Wkt, Kf, nullptr, M, N, K, 0);
  gemm_bf16_wmma<<<gemmGrid, 256, 0, stream>>>(kdb, Wvt, Vf, nullptr, M, N, K, 0);
  gemm_bf16_wmma<<<gemmGrid, 256, 0, stream>>>(qdb, Wgt, G,  bg,      M, N, K, 1);

  // 3) inception depthwise conv; q scaled by D^-0.5; V transposed for PV WMMA
  dwconv_inception<<<EB, 256, 0, stream>>>(Qf, qcw, qcb, Qb, 0.125f, 0);
  dwconv_inception<<<EB, 256, 0, stream>>>(Kf, kcw, kcb, Kb, 1.0f,   0);
  dwconv_inception<<<EB, 256, 0, stream>>>(Vf, vcw, vcb, Vbt, 1.0f,  1);

  // 4) fused flash-style attention: 64 (s,h) pairs x 8 query blocks
  attn_fused<<<512, 256, 0, stream>>>(Qb, Kb, Vbt, bias, k_mask, Of);

  // 5) gating + output projection
  gate_mul<<<EB, 256, 0, stream>>>(Of, G, OGb, NELT);
  gemm_bf16_wmma<<<gemmGrid, 256, 0, stream>>>(OGb, Wot, out, bo, M, N, K, 2);
}
